// ptBEVnet_38225208934760
// MI455X (gfx1250) — matrix operand order repaired and checked by reading, not compiled, not run on hardware
//
#include <hip/hip_runtime.h>
#include <hip/hip_bf16.h>

// ---------------------------------------------------------------------------
// ptBEVnet for MI455X (gfx1250, wave32).
// Pipeline (all on `stream`, graph-capturable):
//   0. memset stats/occ/seg, pack W2..W5 into WMMA-B lane layout (f16)
//   1. input BN0 stats -> scale0/shift0
//   2. layer1 (7->64, VALU) stores x1 f16 pre-BN, accumulates BN1 stats
//   3. bn_relu_apply(x1) ; gemm 64->128  (WMMA, MT=4, double-buffered K)
//   4. bn_relu_apply(x2) ; gemm 128->256 (WMMA, MT=4, double-buffered K)
//   5. bn_relu_apply(x3) ; gemm 256->512 (MT=2) -> ordered-u32 atomicMax seg
//   6. seg_decode (u32->f16) ; head gemm 512->32 (MT=4), occ-masked -> d_out
// GEMM K-loops are software-pipelined: k-step kt+1's A/B fragments are
// loaded before the WMMAs of kt, so waits are partial (loads overlap XDL).
// ---------------------------------------------------------------------------

typedef __attribute__((ext_vector_type(16))) _Float16 v16h;
typedef __attribute__((ext_vector_type(8)))  _Float16 v8h;
typedef __attribute__((ext_vector_type(4)))  _Float16 v4h;
typedef __attribute__((ext_vector_type(8)))  float    v8f;
typedef __attribute__((ext_vector_type(4)))  unsigned int v4u;

static constexpr int kGX   = 480;
static constexpr int kGY   = 360;
static constexpr int kNSEG = kGX * kGY;   // 172800
static constexpr int kNH   = 32;
static constexpr int kB    = 2;
static constexpr int kN    = 120000;
static constexpr int kF    = 7;
static constexpr float kEPS = 1e-5f;

// ----------------------------- small helpers -------------------------------

__device__ __forceinline__ unsigned int seg_encode(float v) {
  unsigned int u = __float_as_uint(v);
  return (u & 0x80000000u) ? ~u : (u | 0x80000000u);   // monotonic in unsigned
}
__device__ __forceinline__ float seg_decode_f(unsigned int u) {
  if (u == 0u) return 0.f;                             // empty sentinel
  unsigned int bits = (u & 0x80000000u) ? (u & 0x7FFFFFFFu) : ~u;
  return __uint_as_float(bits);
}

// Assemble CDNA5 16-bit A-operand fragment: halves kb..kb+7 and kb+16..kb+23.
__device__ __forceinline__ v16h load_a16(const _Float16* __restrict__ p) {
  union { v16h v; struct { v8h lo; v8h hi; } s; } u;
  u.s.lo = *(const v8h*)p;
  u.s.hi = *(const v8h*)(p + 16);
  return u.v;
}

// Pack row-major f32 W[K][Nout] into WMMA-B f16 panels:
//   Wp[((nt*nkt + kt)*32 + lane)*16 + i] = W[kt*32 + ((i>>3)<<4) + ((lane>>4)<<3) + (i&7)][nt*16 + (lane&15)]
__global__ void pack_w(const float* __restrict__ W, _Float16* __restrict__ Wp,
                       int K, int Nout) {
  int tid = blockIdx.x * blockDim.x + threadIdx.x;
  if (tid >= K * Nout) return;
  int i    = tid & 15;
  int lane = (tid >> 4) & 31;
  int tile = tid >> 9;
  int nkt  = K >> 5;
  int kt   = tile % nkt;
  int nt   = tile / nkt;
  int n = nt * 16 + (lane & 15);
  int k = kt * 32 + ((i >> 3) << 4) + ((lane >> 4) << 3) + (i & 7);
  Wp[tid] = (_Float16)W[(size_t)k * Nout + n];
}

// Flatten voxel ids, mark occupancy.
__global__ void ids_occ(const int* __restrict__ xy, int* __restrict__ ids,
                        unsigned char* __restrict__ occ) {
  int t = blockIdx.x * blockDim.x + threadIdx.x;
  if (t >= kB * kN) return;
  int id = xy[2 * t] * kGY + xy[2 * t + 1];
  ids[t] = id;
  int b = t / kN;
  occ[(size_t)b * kNSEG + id] = 1;
}

// Per-sample per-feature sum / sumsq of the raw input (for BN0).
__global__ void input_stats(const float* __restrict__ fea,
                            float* __restrict__ sum, float* __restrict__ sq) {
  int b = blockIdx.y;
  __shared__ float ls[kF], lq[kF];
  if (threadIdx.x < kF) { ls[threadIdx.x] = 0.f; lq[threadIdx.x] = 0.f; }
  __syncthreads();
  float s[kF] = {}, q[kF] = {};
  for (int p = blockIdx.x * blockDim.x + threadIdx.x; p < kN;
       p += gridDim.x * blockDim.x) {
    const float* f = fea + ((size_t)b * kN + p) * kF;
#pragma unroll
    for (int k = 0; k < kF; ++k) { float v = f[k]; s[k] += v; q[k] += v * v; }
  }
#pragma unroll
  for (int k = 0; k < kF; ++k) { atomicAdd(&ls[k], s[k]); atomicAdd(&lq[k], q[k]); }
  __syncthreads();
  if (threadIdx.x < kF) {
    atomicAdd(&sum[b * kF + threadIdx.x], ls[threadIdx.x]);
    atomicAdd(&sq [b * kF + threadIdx.x], lq[threadIdx.x]);
  }
}

// Fold BN stats into affine scale/shift:  bn(x) = scale*x + shift.
__global__ void bn_prep(const float* __restrict__ sum, const float* __restrict__ sq,
                        const float* __restrict__ g, const float* __restrict__ bb,
                        float* __restrict__ scale, float* __restrict__ shift,
                        int K, float invN) {
  int b = blockIdx.y;
  int k = threadIdx.x;
  if (k >= K) return;
  float m = sum[b * K + k] * invN;
  float v = sq[b * K + k] * invN - m * m;
  float a = g[k] * rsqrtf(v + kEPS);
  scale[b * K + k] = a;
  shift[b * K + k] = bb[k] - a * m;
}

// In-place elementwise BN + ReLU over pre-BN f16 activations (8 halfs/thread).
template <int K>
__global__ void __launch_bounds__(256)
bn_relu_apply(_Float16* __restrict__ x, const float* __restrict__ scale,
              const float* __restrict__ shift) {
  int b = blockIdx.y;
  __shared__ float sS[K], sH[K];
  for (int i = threadIdx.x; i < K; i += 256) {
    sS[i] = scale[b * K + i];
    sH[i] = shift[b * K + i];
  }
  __syncthreads();
  size_t base = ((size_t)blockIdx.x * 256 + threadIdx.x) * 8;
  if (base >= (size_t)kN * K) return;
  int k0 = (int)(base & (size_t)(K - 1));   // K is a power of two
  _Float16* p = x + (size_t)b * kN * K + base;
  v8h v = *(const v8h*)p;
#pragma unroll
  for (int i = 0; i < 8; ++i) {
    float f = fmaf((float)v[i], sS[k0 + i], sH[k0 + i]);
    v[i] = (_Float16)fmaxf(f, 0.f);
  }
  *(v8h*)p = v;
}

// Layer 1: y = bn0(x) @ W1 + b1 (K=7 -> VALU), store f16, accumulate BN1 stats.
__global__ void __launch_bounds__(256)
layer1(const float* __restrict__ fea, const float* __restrict__ scale0,
       const float* __restrict__ shift0, const float* __restrict__ W1,
       const float* __restrict__ b1, _Float16* __restrict__ x1,
       float* __restrict__ sum1, float* __restrict__ sq1) {
  int b = blockIdx.y;
  __shared__ float sW[kF * 64], sB[64], ssum[64], ssq[64];
  for (int i = threadIdx.x; i < kF * 64; i += blockDim.x) sW[i] = W1[i];
  if (threadIdx.x < 64) {
    sB[threadIdx.x] = b1[threadIdx.x];
    ssum[threadIdx.x] = 0.f; ssq[threadIdx.x] = 0.f;
  }
  __syncthreads();
  int p = blockIdx.x * blockDim.x + threadIdx.x;
  if (p < kN) {
    float xb[kF];
    const float* f = fea + ((size_t)b * kN + p) * kF;
#pragma unroll
    for (int k = 0; k < kF; ++k)
      xb[k] = fmaf(f[k], scale0[b * kF + k], shift0[b * kF + k]);
    _Float16* yr = x1 + ((size_t)b * kN + p) * 64;
#pragma unroll 4
    for (int c = 0; c < 64; ++c) {
      float acc = sB[c];
#pragma unroll
      for (int k = 0; k < kF; ++k) acc = fmaf(xb[k], sW[k * 64 + c], acc);
      yr[c] = (_Float16)acc;
      atomicAdd(&ssum[c], acc);
      atomicAdd(&ssq[c], acc * acc);
    }
  }
  __syncthreads();
  if (threadIdx.x < 64) {
    atomicAdd(&sum1[b * 64 + threadIdx.x], ssum[threadIdx.x]);
    atomicAdd(&sq1 [b * 64 + threadIdx.x], ssq[threadIdx.x]);
  }
}

// Pure WMMA GEMM with double-buffered K pipeline: y = x @ Wp + bias;
// store f16 pre-BN y; accumulate stats. Each wave: MT M-tiles x NTPW N-tiles.
template <int K, int NOUT, int NWAVES, int MT>
__global__ void __launch_bounds__(NWAVES * 32)
gemm_layer(const _Float16* __restrict__ xin, const _Float16* __restrict__ Wp,
           const float* __restrict__ bias, _Float16* __restrict__ yout,
           float* __restrict__ statSum, float* __restrict__ statSq) {
  constexpr int NKT  = K / 32;          // even for all instantiations
  constexpr int NNT  = NOUT / 16;
  constexpr int NTPW = NNT / NWAVES;
  int b    = blockIdx.y;
  int m0   = blockIdx.x * 16 * MT;
  int lane = threadIdx.x & 31, wave = threadIdx.x >> 5;
  int kA0  = (lane < 16) ? 0 : 8;
  const _Float16* xrow[MT];
#pragma unroll
  for (int t = 0; t < MT; ++t)
    xrow[t] = xin + ((size_t)b * kN + m0 + t * 16 + (lane & 15)) * K + kA0;

  auto loadA = [&](int kt, v16h* dst) {
#pragma unroll
    for (int t = 0; t < MT; ++t) dst[t] = load_a16(xrow[t] + kt * 32);
  };
  auto loadB = [&](int kt, v16h* dst) {
#pragma unroll
    for (int j = 0; j < NTPW; ++j) {
      int nt = wave + j * NWAVES;
      dst[j] = *(const v16h*)(Wp + ((size_t)(nt * NKT + kt) * 32 + lane) * 16);
    }
  };

  v8f acc[NTPW][MT] = {};
  v16h aP[MT], bP[NTPW], aQ[MT], bQ[NTPW];
  loadA(0, aP); loadB(0, bP);
#pragma unroll 1
  for (int kt = 0; kt < NKT; kt += 2) {
    loadA(kt + 1, aQ); loadB(kt + 1, bQ);         // prefetch odd step
#pragma unroll
    for (int j = 0; j < NTPW; ++j)
#pragma unroll
      for (int t = 0; t < MT; ++t)
        acc[j][t] = __builtin_amdgcn_wmma_f32_16x16x32_f16(
            false, aP[t], false, bP[j], (short)0, acc[j][t], false, false);
    int kt2 = (kt + 2 < NKT) ? kt + 2 : kt + 1;   // clamped (dead on last step)
    loadA(kt2, aP); loadB(kt2, bP);               // prefetch next even step
#pragma unroll
    for (int j = 0; j < NTPW; ++j)
#pragma unroll
      for (int t = 0; t < MT; ++t)
        acc[j][t] = __builtin_amdgcn_wmma_f32_16x16x32_f16(
            false, aQ[t], false, bQ[j], (short)0, acc[j][t], false, false);
  }

  int roff = (lane >> 4) << 3;
#pragma unroll
  for (int j = 0; j < NTPW; ++j) {
    int c  = (wave + j * NWAVES) * 16 + (lane & 15);
    float bv = bias[c];
    float s = 0.f, q = 0.f;
#pragma unroll
    for (int t = 0; t < MT; ++t) {
#pragma unroll
      for (int r = 0; r < 8; ++r) {
        float v = acc[j][t][r] + bv;
        yout[((size_t)b * kN + m0 + t * 16 + roff + r) * NOUT + c] = (_Float16)v;
        s += v; q += v * v;
      }
    }
    atomicAdd(&statSum[b * NOUT + c], s);
    atomicAdd(&statSq [b * NOUT + c], q);
  }
}

// x4 = x3bn @ W4 + b4, then per-voxel atomic max scatter (ordered u32).
template <int K, int NOUT, int NWAVES, int MT>
__global__ void __launch_bounds__(NWAVES * 32)
gemm_scatter(const _Float16* __restrict__ xin, const _Float16* __restrict__ Wp,
             const float* __restrict__ bias, const int* __restrict__ ids,
             unsigned int* __restrict__ seg) {
  constexpr int NKT  = K / 32;
  constexpr int NNT  = NOUT / 16;
  constexpr int NTPW = NNT / NWAVES;
  int b    = blockIdx.y;
  int m0   = blockIdx.x * 16 * MT;
  int lane = threadIdx.x & 31, wave = threadIdx.x >> 5;
  int kA0  = (lane < 16) ? 0 : 8;
  const _Float16* xrow[MT];
#pragma unroll
  for (int t = 0; t < MT; ++t)
    xrow[t] = xin + ((size_t)b * kN + m0 + t * 16 + (lane & 15)) * K + kA0;

  auto loadA = [&](int kt, v16h* dst) {
#pragma unroll
    for (int t = 0; t < MT; ++t) dst[t] = load_a16(xrow[t] + kt * 32);
  };
  auto loadB = [&](int kt, v16h* dst) {
#pragma unroll
    for (int j = 0; j < NTPW; ++j) {
      int nt = wave + j * NWAVES;
      dst[j] = *(const v16h*)(Wp + ((size_t)(nt * NKT + kt) * 32 + lane) * 16);
    }
  };

  v8f acc[NTPW][MT] = {};
  v16h aP[MT], bP[NTPW], aQ[MT], bQ[NTPW];
  loadA(0, aP); loadB(0, bP);
#pragma unroll 1
  for (int kt = 0; kt < NKT; kt += 2) {
    loadA(kt + 1, aQ); loadB(kt + 1, bQ);
#pragma unroll
    for (int j = 0; j < NTPW; ++j)
#pragma unroll
      for (int t = 0; t < MT; ++t)
        acc[j][t] = __builtin_amdgcn_wmma_f32_16x16x32_f16(
            false, aP[t], false, bP[j], (short)0, acc[j][t], false, false);
    int kt2 = (kt + 2 < NKT) ? kt + 2 : kt + 1;
    loadA(kt2, aP); loadB(kt2, bP);
#pragma unroll
    for (int j = 0; j < NTPW; ++j)
#pragma unroll
      for (int t = 0; t < MT; ++t)
        acc[j][t] = __builtin_amdgcn_wmma_f32_16x16x32_f16(
            false, aQ[t], false, bQ[j], (short)0, acc[j][t], false, false);
  }

  int roff = (lane >> 4) << 3;
  int idr[MT][8];
#pragma unroll
  for (int t = 0; t < MT; ++t)
#pragma unroll
    for (int r = 0; r < 8; ++r)
      idr[t][r] = ids[(size_t)b * kN + m0 + t * 16 + roff + r];
#pragma unroll
  for (int j = 0; j < NTPW; ++j) {
    int c  = (wave + j * NWAVES) * 16 + (lane & 15);
    float bv = bias[c];
#pragma unroll
    for (int t = 0; t < MT; ++t) {
#pragma unroll
      for (int r = 0; r < 8; ++r) {
        unsigned int enc = seg_encode(acc[j][t][r] + bv);
        atomicMax(&seg[((size_t)b * kNSEG + idr[t][r]) * NOUT + c], enc);
      }
    }
  }
}

// Elementwise ordered-u32 -> f16 decode of the voxel max buffer (4/thread).
__global__ void __launch_bounds__(256)
seg_decode_kernel(const unsigned int* __restrict__ seg,
                  _Float16* __restrict__ seg16) {
  size_t t = ((size_t)blockIdx.x * 256 + threadIdx.x) * 4;
  if (t >= (size_t)kB * kNSEG * 512) return;
  v4u u = *(const v4u*)(seg + t);
  v4h o;
#pragma unroll
  for (int i = 0; i < 4; ++i) o[i] = (_Float16)seg_decode_f(u[i]);
  *(v4h*)(seg16 + t) = o;
}

// Head: nh = occ ? relu(seg16 @ W5 + b5) : 0, written as [B][NH][GX*GY].
template <int MT>
__global__ void __launch_bounds__(64)
gemm_head(const _Float16* __restrict__ a16, const _Float16* __restrict__ Wp,
          const float* __restrict__ b5, const unsigned char* __restrict__ occ,
          float* __restrict__ out) {
  constexpr int K = 512, NKT = K / 32;
  int b    = blockIdx.y;
  int m0   = blockIdx.x * 16 * MT;
  int lane = threadIdx.x & 31, wave = threadIdx.x >> 5;   // wave == n-tile
  int kA0  = (lane < 16) ? 0 : 8;
  const _Float16* xrow[MT];
#pragma unroll
  for (int t = 0; t < MT; ++t)
    xrow[t] = a16 + ((size_t)b * kNSEG + m0 + t * 16 + (lane & 15)) * K + kA0;

  auto loadA = [&](int kt, v16h* dst) {
#pragma unroll
    for (int t = 0; t < MT; ++t) dst[t] = load_a16(xrow[t] + kt * 32);
  };
  auto loadB = [&](int kt) {
    return *(const v16h*)(Wp + ((size_t)(wave * NKT + kt) * 32 + lane) * 16);
  };

  v8f acc[MT] = {};
  v16h aP[MT], aQ[MT];
  v16h bP = loadB(0), bQ;
  loadA(0, aP);
#pragma unroll 1
  for (int kt = 0; kt < NKT; kt += 2) {
    loadA(kt + 1, aQ); bQ = loadB(kt + 1);
#pragma unroll
    for (int t = 0; t < MT; ++t)
      acc[t] = __builtin_amdgcn_wmma_f32_16x16x32_f16(
          false, aP[t], false, bP, (short)0, acc[t], false, false);
    int kt2 = (kt + 2 < NKT) ? kt + 2 : kt + 1;
    loadA(kt2, aP); bP = loadB(kt2);
#pragma unroll
    for (int t = 0; t < MT; ++t)
      acc[t] = __builtin_amdgcn_wmma_f32_16x16x32_f16(
          false, aQ[t], false, bQ, (short)0, acc[t], false, false);
  }

  int c    = wave * 16 + (lane & 15);
  int roff = (lane >> 4) << 3;
  float bv = b5[c];
#pragma unroll
  for (int t = 0; t < MT; ++t) {
#pragma unroll
    for (int r = 0; r < 8; ++r) {
      int sid = m0 + t * 16 + roff + r;
      float v = fmaxf(acc[t][r] + bv, 0.f);
      float o = occ[(size_t)b * kNSEG + sid] ? v : 0.f;
      out[((size_t)b * kNH + c) * kNSEG + sid] = o;
    }
  }
}

// ---------------------------------------------------------------------------

extern "C" void kernel_launch(void* const* d_in, const int* in_sizes, int n_in,
                              void* d_out, int out_size, void* d_ws, size_t ws_size,
                              hipStream_t stream) {
  (void)in_sizes; (void)n_in; (void)out_size; (void)ws_size;
  const float* pt_fea = (const float*)d_in[0];
  const int*   xy_ind = (const int*)  d_in[1];
  const float* bn0_g  = (const float*)d_in[2];
  const float* bn0_b  = (const float*)d_in[3];
  const float* W1 = (const float*)d_in[4];
  const float* b1 = (const float*)d_in[5];
  const float* bn1_g = (const float*)d_in[6];
  const float* bn1_b = (const float*)d_in[7];
  const float* W2 = (const float*)d_in[8];
  const float* b2 = (const float*)d_in[9];
  const float* bn2_g = (const float*)d_in[10];
  const float* bn2_b = (const float*)d_in[11];
  const float* W3 = (const float*)d_in[12];
  const float* b3 = (const float*)d_in[13];
  const float* bn3_g = (const float*)d_in[14];
  const float* bn3_b = (const float*)d_in[15];
  const float* W4 = (const float*)d_in[16];
  const float* b4 = (const float*)d_in[17];
  const float* W5 = (const float*)d_in[18];
  const float* b5 = (const float*)d_in[19];
  float* out = (float*)d_out;

  // ---- workspace carve (256B aligned) ----
  char* base = (char*)d_ws;
  size_t off = 0;
  auto take = [&](size_t bytes) -> void* {
    void* p = base + off;
    off = (off + bytes + 255) & ~(size_t)255;
    return p;
  };
  const int statFloats = kB * (kF + 64 + 128 + 256) * 2;  // 1820
  float* statsBase = (float*)take(statFloats * sizeof(float));
  float* sum0 = statsBase;          float* sq0 = sum0 + kB * kF;
  float* sum1 = sq0 + kB * kF;      float* sq1 = sum1 + kB * 64;
  float* sum2 = sq1 + kB * 64;      float* sq2 = sum2 + kB * 128;
  float* sum3 = sq2 + kB * 128;     float* sq3 = sum3 + kB * 256;

  float* scale0 = (float*)take(kB * kF  * 4); float* shift0 = (float*)take(kB * kF  * 4);
  float* scale1 = (float*)take(kB * 64  * 4); float* shift1 = (float*)take(kB * 64  * 4);
  float* scale2 = (float*)take(kB * 128 * 4); float* shift2 = (float*)take(kB * 128 * 4);
  float* scale3 = (float*)take(kB * 256 * 4); float* shift3 = (float*)take(kB * 256 * 4);

  _Float16* x1  = (_Float16*)take((size_t)kB * kN * 64  * 2);
  _Float16* x2  = (_Float16*)take((size_t)kB * kN * 128 * 2);
  _Float16* x3  = (_Float16*)take((size_t)kB * kN * 256 * 2);
  _Float16* Wp2 = (_Float16*)take(64  * 128 * 2);
  _Float16* Wp3 = (_Float16*)take(128 * 256 * 2);
  _Float16* Wp4 = (_Float16*)take(256 * 512 * 2);
  _Float16* Wp5 = (_Float16*)take(512 * 32  * 2);
  int* ids = (int*)take((size_t)kB * kN * 4);
  unsigned char* occ = (unsigned char*)take((size_t)kB * kNSEG);
  unsigned int* seg  = (unsigned int*)take((size_t)kB * kNSEG * 512 * 4);
  _Float16* seg16    = (_Float16*)take((size_t)kB * kNSEG * 512 * 2);

  // ---- init ----
  hipMemsetAsync(statsBase, 0, statFloats * sizeof(float), stream);
  hipMemsetAsync(occ, 0, (size_t)kB * kNSEG, stream);
  hipMemsetAsync(seg, 0, (size_t)kB * kNSEG * 512 * 4, stream);

  // ---- weight packing + ids ----
  pack_w<<<dim3((64 * 128 + 255) / 256), 256, 0, stream>>>(W2, Wp2, 64, 128);
  pack_w<<<dim3((128 * 256 + 255) / 256), 256, 0, stream>>>(W3, Wp3, 128, 256);
  pack_w<<<dim3((256 * 512 + 255) / 256), 256, 0, stream>>>(W4, Wp4, 256, 512);
  pack_w<<<dim3((512 * 32 + 255) / 256), 256, 0, stream>>>(W5, Wp5, 512, 32);
  ids_occ<<<dim3((kB * kN + 255) / 256), 256, 0, stream>>>(xy_ind, ids, occ);

  const float invN = 1.f / (float)kN;

  // ---- BN0 + layer1 ----
  input_stats<<<dim3(64, kB), 256, 0, stream>>>(pt_fea, sum0, sq0);
  bn_prep<<<dim3(1, kB), 256, 0, stream>>>(sum0, sq0, bn0_g, bn0_b, scale0, shift0, kF, invN);
  layer1<<<dim3((kN + 255) / 256, kB), 256, 0, stream>>>(pt_fea, scale0, shift0,
                                                         W1, b1, x1, sum1, sq1);
  // ---- layer2: 64 -> 128 ----
  bn_prep<<<dim3(1, kB), 256, 0, stream>>>(sum1, sq1, bn1_g, bn1_b, scale1, shift1, 64, invN);
  bn_relu_apply<64><<<dim3(kN * 64 / 8 / 256, kB), 256, 0, stream>>>(x1, scale1, shift1);
  gemm_layer<64, 128, 8, 4><<<dim3(kN / 64, kB), 256, 0, stream>>>(
      x1, Wp2, b2, x2, sum2, sq2);
  // ---- layer3: 128 -> 256 ----
  bn_prep<<<dim3(1, kB), 256, 0, stream>>>(sum2, sq2, bn2_g, bn2_b, scale2, shift2, 128, invN);
  bn_relu_apply<128><<<dim3(kN * 128 / 8 / 256, kB), 256, 0, stream>>>(x2, scale2, shift2);
  gemm_layer<128, 256, 8, 4><<<dim3(kN / 64, kB), 256, 0, stream>>>(
      x2, Wp3, b3, x3, sum3, sq3);
  // ---- layer4: 256 -> 512 + voxel max scatter ----
  bn_prep<<<dim3(1, kB), 256, 0, stream>>>(sum3, sq3, bn3_g, bn3_b, scale3, shift3, 256, invN);
  bn_relu_apply<256><<<dim3(kN * 256 / 8 / 256, kB), 256, 0, stream>>>(x3, scale3, shift3);
  gemm_scatter<256, 512, 8, 2><<<dim3(kN / 32, kB), 256, 0, stream>>>(
      x3, Wp4, b4, ids, seg);
  // ---- voxel head: 512 -> 32 ----
  seg_decode_kernel<<<dim3((unsigned)((size_t)kB * kNSEG * 512 / 4 / 256)), 256, 0,
                      stream>>>(seg, seg16);
  gemm_head<4><<<dim3(kNSEG / 64, kB), 64, 0, stream>>>(seg16, Wp5, b5, occ, out);
}
